// PointerNetwork_6519760355909
// MI455X (gfx1250) — matrix operand order repaired and checked by reading, not compiled, not run on hardware
//
#include <hip/hip_runtime.h>
#include <stdint.h>

// Problem constants (match reference)
#define Bm   256
#define Nn   128
#define Ee   256
#define Hh   512
#define G4   2048          // 4*H
#define NP1  129           // N+1
#define Tt   129           // decoder steps
#define NEGV (-1e30f)

// LDS panel row strides (bf16 elements), padded +8 to break bank conflicts
#define XS   (Ee + 8)      // 264
#define HS   (Hh + 8)      // 520

typedef __bf16 bf16_t;
typedef __attribute__((ext_vector_type(16))) __bf16 v16bf;
typedef __attribute__((ext_vector_type(8)))  float  v8f;

// ---------------- WMMA fragment loaders (CDNA5 16x16x32 bf16 layouts) -------
// A: 16x32 (MxK). lanes 0-15 -> M=lane, lanes 16-31 -> M=lane-16.
// VGPR v (2 bf16 each): v<4 -> K = half*8 + 2v,(+1); v>=4 -> K = 16 + half*8 + 2(v-4),(+1)
__device__ __forceinline__ v16bf load_a_lds(const bf16_t* base, int stride, int k0) {
    int lane = threadIdx.x & 31;
    int half = lane >> 4;
    int m    = lane & 15;
    const bf16_t* row = base + m * stride + k0;
    v16bf a;
    uint32_t* ap = (uint32_t*)&a;
#pragma unroll
    for (int v = 0; v < 8; ++v) {
        int kk = (v < 4) ? (half * 8 + 2 * v) : (16 + half * 8 + 2 * (v - 4));
        ap[v] = *(const uint32_t*)(row + kk);   // ds_load_b32
    }
    return a;
}

// Same A layout but converting fp32 global source -> bf16 on the fly.
__device__ __forceinline__ v16bf load_a_f32(const float* base, int row0, int ldx, int k0) {
    int lane = threadIdx.x & 31;
    int half = lane >> 4;
    int m    = lane & 15;
    const float* row = base + (size_t)(row0 + m) * ldx + k0;
    v16bf a;
#pragma unroll
    for (int v = 0; v < 8; ++v) {
        int kk = (v < 4) ? (half * 8 + 2 * v) : (16 + half * 8 + 2 * (v - 4));
        float2 p = *(const float2*)(row + kk);
        a[2 * v]     = (__bf16)p.x;
        a[2 * v + 1] = (__bf16)p.y;
    }
    return a;
}

// B: 32x16 (KxN) with B[k,n] = W[n0+n, k0+k]; W row-major [Nout, K] bf16.
// VGPR v (2 bf16 each): v<4 -> K = half*8 + 2v,(+1); v>=4 -> K = 16 + half*8 + 2(v-4),(+1)
__device__ __forceinline__ v16bf load_b_w(const bf16_t* W, int n0, int ldw, int k0) {
    int lane = threadIdx.x & 31;
    int half = lane >> 4;
    int n    = lane & 15;
    const bf16_t* row = W + (size_t)(n0 + n) * ldw + k0;
    v16bf b;
    uint32_t* bp = (uint32_t*)&b;
#pragma unroll
    for (int v = 0; v < 8; ++v) {
        int kk = (v < 4) ? (half * 8 + 2 * v) : (16 + half * 8 + 2 * (v - 4));
        bp[v] = *(const uint32_t*)(row + kk);
    }
    return b;
}

// issue one 16B async global->LDS copy (tracked by ASYNCcnt)
__device__ __forceinline__ void async_copy_b128(uint32_t lds_off, const void* gptr) {
    unsigned long long ga = (unsigned long long)(uintptr_t)gptr;
    asm volatile("global_load_async_to_lds_b128 %0, %1, off"
                 :: "v"(lds_off), "v"(ga) : "memory");
}

// ---------------- GEMM kernels ----------------------------------------------
// node_embeds = adj(B*N, N) x W_embed^T(N, E) + b_embed ; output bf16 (one-shot)
__global__ void embed_gemm(const float* __restrict__ adj,
                           const bf16_t* __restrict__ Wemb,  // [E, N] bf16
                           const float* __restrict__ bemb,
                           bf16_t* __restrict__ emb)         // [B*N, E] bf16
{
    int wave = (blockIdx.x * blockDim.x + threadIdx.x) >> 5;
    int lane = threadIdx.x & 31;
    const int ntiles = Ee / 16;            // 16
    int mtile = wave / ntiles;             // 0..2047
    int ntile = wave % ntiles;
    int row0 = mtile * 16, n0 = ntile * 16;
    v8f acc = {};
#pragma unroll
    for (int k0 = 0; k0 < Nn; k0 += 32) {
        v16bf a = load_a_f32(adj, row0, Nn, k0);
        v16bf b = load_b_w(Wemb, n0, Nn, k0);
        acc = __builtin_amdgcn_wmma_f32_16x16x32_bf16(false, a, false, b,
                                                      (short)0, acc, false, false);
    }
    int half = lane >> 4;
    int ncol = n0 + (lane & 15);
    float bias = bemb[ncol];
#pragma unroll
    for (int i = 0; i < 8; ++i) {
        int mrow = row0 + i + 8 * half;
        emb[(size_t)mrow * Ee + ncol] = (__bf16)(acc[i] + bias);
    }
}

// One LSTM step pre-activation: z = x(B,E)*Wih^T + h(B,H)*Whh^T + bih + bhh
// Block = 8 waves sharing one mtile (16 batch rows): the 16x(E+H) A panel is
// staged once into LDS via async global->LDS copies; B (weights) stream from L2.
__global__ void lstm_gemm_step(const bf16_t* __restrict__ x, int ldx,
                               const bf16_t* __restrict__ h,
                               const bf16_t* __restrict__ Wih,   // [4H, E] bf16
                               const bf16_t* __restrict__ Whh,   // [4H, H] bf16
                               const float* __restrict__ bih,
                               const float* __restrict__ bhh,
                               float* __restrict__ z)            // [B, 4H]
{
    __shared__ bf16_t sax[16 * XS];    // 16 rows of x   (E cols + pad)
    __shared__ bf16_t sah[16 * HS];    // 16 rows of h   (H cols + pad)

    int tid   = threadIdx.x;
    int lane  = tid & 31;
    int wave  = tid >> 5;              // 0..7
    int blk   = blockIdx.x;            // 0..255
    int mtile = blk >> 4;              // 0..15  (16 batch-row tiles)
    int ngrp  = blk & 15;              // 0..15  (groups of 8 ntiles)
    int ntile = ngrp * 8 + wave;       // 0..127
    int row0  = mtile * 16, n0 = ntile * 16;

    // ---- stage A panel into LDS (async, ASYNCcnt-tracked) ----
    // x: 16 rows * 32 chunks of 16B ; h: 16 rows * 64 chunks of 16B
    for (int c = tid; c < 16 * 32; c += 256) {
        int r = c >> 5, q = c & 31;
        uint32_t l = (uint32_t)(uintptr_t)&sax[r * XS + q * 8];
        async_copy_b128(l, x + (size_t)(row0 + r) * ldx + q * 8);
    }
    for (int c = tid; c < 16 * 64; c += 256) {
        int r = c >> 6, q = c & 63;
        uint32_t l = (uint32_t)(uintptr_t)&sah[r * HS + q * 8];
        async_copy_b128(l, h + (size_t)(row0 + r) * Hh + q * 8);
    }
    asm volatile("s_wait_asynccnt 0x0" ::: "memory");
    __syncthreads();

    // ---- K loops: A from LDS, B from global (L2-resident weights) ----
    v8f acc = {};
#pragma unroll
    for (int k0 = 0; k0 < Ee; k0 += 32) {
        v16bf a = load_a_lds(sax, XS, k0);
        v16bf b = load_b_w(Wih, n0, Ee, k0);
        acc = __builtin_amdgcn_wmma_f32_16x16x32_bf16(false, a, false, b,
                                                      (short)0, acc, false, false);
    }
#pragma unroll
    for (int k0 = 0; k0 < Hh; k0 += 32) {
        v16bf a = load_a_lds(sah, HS, k0);
        v16bf b = load_b_w(Whh, n0, Hh, k0);
        acc = __builtin_amdgcn_wmma_f32_16x16x32_bf16(false, a, false, b,
                                                      (short)0, acc, false, false);
    }
    int half = lane >> 4;
    int ncol = n0 + (lane & 15);
    float bias = bih[ncol] + bhh[ncol];
#pragma unroll
    for (int i = 0; i < 8; ++i) {
        int mrow = row0 + i + 8 * half;
        z[(size_t)mrow * G4 + ncol] = acc[i] + bias;
    }
}

// ---------------- elementwise / reduction kernels ---------------------------
__global__ void lstm_gate(const float* __restrict__ z, float* __restrict__ c,
                          float* __restrict__ h, bf16_t* __restrict__ hbf,
                          float* __restrict__ ext_enc, int t)
{
    int idx = blockIdx.x * blockDim.x + threadIdx.x;
    if (idx >= Bm * Hh) return;
    int b = idx / Hh, j = idx % Hh;
    const float* zr = z + (size_t)b * G4;
    float zi = zr[j], zfv = zr[Hh + j], zg = zr[2 * Hh + j], zo = zr[3 * Hh + j];
    float si = 1.f / (1.f + expf(-zi));
    float sf = 1.f / (1.f + expf(-zfv));
    float so = 1.f / (1.f + expf(-zo));
    float cn = sf * c[idx] + si * tanhf(zg);
    float hn = so * tanhf(cn);
    c[idx] = cn;
    h[idx] = hn;
    hbf[idx] = (__bf16)hn;
    if (ext_enc)   // encoder: also record into extended memory [B, N+1, H]
        ext_enc[((size_t)b * NP1 + t) * Hh + j] = hn;
}

// logits[b,j] = dot(ext_enc[b,j,:], h[b,:]) ; one wave32 per (b,j)
__global__ void attn_logits(const float* __restrict__ ext_enc,
                            const float* __restrict__ h,
                            float* __restrict__ logits)
{
    int wave = (blockIdx.x * blockDim.x + threadIdx.x) >> 5;
    int lane = threadIdx.x & 31;
    if (wave >= Bm * NP1) return;
    int b = wave / NP1, j = wave % NP1;
    const float* e  = ext_enc + ((size_t)b * NP1 + j) * Hh;
    const float* hb = h + (size_t)b * Hh;
    float s = 0.f;
#pragma unroll
    for (int k = lane; k < Hh; k += 32) s += e[k] * hb[k];
#pragma unroll
    for (int off = 16; off > 0; off >>= 1) s += __shfl_xor(s, off, 32);
    if (lane == 0) logits[(size_t)b * NP1 + j] = s;
}

// masked log-softmax CE for step t; one wave32 per batch row; updates mask
__global__ void ce_loss_step(const float* __restrict__ logits,
                             unsigned char* __restrict__ mask,
                             const int* __restrict__ tgt, int t,
                             float* __restrict__ loss)
{
    int b = blockIdx.x;
    int lane = threadIdx.x;
    const float* lg = logits + (size_t)b * NP1;
    unsigned char* mk = mask + (size_t)b * NP1;
    float vals[5];
    float mx = -3.4e38f;
#pragma unroll
    for (int i = 0; i < 5; ++i) {
        int j = lane + 32 * i;
        float v = -3.4e38f;
        if (j < NP1) v = mk[j] ? NEGV : lg[j];
        vals[i] = v;
        mx = fmaxf(mx, v);
    }
#pragma unroll
    for (int off = 16; off > 0; off >>= 1) mx = fmaxf(mx, __shfl_xor(mx, off, 32));
    float se = 0.f;
#pragma unroll
    for (int i = 0; i < 5; ++i) {
        int j = lane + 32 * i;
        if (j < NP1) se += expf(vals[i] - mx);
    }
#pragma unroll
    for (int off = 16; off > 0; off >>= 1) se += __shfl_xor(se, off, 32);
    if (lane == 0) {
        int tj = tgt[(size_t)b * NP1 + t];
        float xt = mk[tj] ? NEGV : lg[tj];
        float logp = (xt - mx) - logf(se);
        atomicAdd(loss, -logp);
        mk[tj] = 1;   // after loss, as in reference
    }
}

// next decoder input: node_embeds[b, tgt[b,t]] or 0 for EOS token
__global__ void gather_dec(const bf16_t* __restrict__ emb,
                           const int* __restrict__ tgt, int t,
                           bf16_t* __restrict__ dec_in)
{
    int i = blockIdx.x * blockDim.x + threadIdx.x;
    if (i >= Bm * Ee) return;
    int b = i / Ee, e = i % Ee;
    int tj = tgt[(size_t)b * NP1 + t];
    dec_in[i] = (tj >= Nn) ? (__bf16)0.f : emb[((size_t)b * Nn + tj) * Ee + e];
}

// ---------------- init / finalize -------------------------------------------
__global__ void f32_to_bf16(const float* __restrict__ s, bf16_t* __restrict__ d, int n) {
    int i = blockIdx.x * blockDim.x + threadIdx.x;
    if (i < n) d[i] = (__bf16)s[i];
}

__global__ void init_state(float* h, bf16_t* hbf, float* c,
                           unsigned char* mask, float* loss) {
    int i = blockIdx.x * blockDim.x + threadIdx.x;
    if (i < Bm * Hh) { h[i] = 0.f; hbf[i] = (__bf16)0.f; c[i] = 0.f; }
    if (i < Bm * NP1) mask[i] = 0;
    if (i == 0) *loss = 0.f;
}

__global__ void init_dec0(const float* __restrict__ dstart, bf16_t* __restrict__ dec_in) {
    int i = blockIdx.x * blockDim.x + threadIdx.x;
    if (i < Bm * Ee) dec_in[i] = (__bf16)dstart[i % Ee];
}

__global__ void init_eos(const float* __restrict__ eos, float* __restrict__ ext_enc) {
    int i = blockIdx.x * blockDim.x + threadIdx.x;
    if (i < Bm * Hh) {
        int b = i / Hh, j = i % Hh;
        ext_enc[((size_t)b * NP1 + Nn) * Hh + j] = eos[j];
    }
}

__global__ void finalize(const float* __restrict__ loss, float* __restrict__ out) {
    if (blockIdx.x == 0 && threadIdx.x == 0) out[0] = loss[0] / (float)(Bm * Tt);
}

// ---------------- host driver -----------------------------------------------
extern "C" void kernel_launch(void* const* d_in, const int* in_sizes, int n_in,
                              void* d_out, int out_size, void* d_ws, size_t ws_size,
                              hipStream_t stream)
{
    (void)in_sizes; (void)n_in; (void)out_size; (void)ws_size;
    const float* adj    = (const float*)d_in[0];
    const int*   tgt    = (const int*)  d_in[1];
    const float* Wemb   = (const float*)d_in[2];
    const float* bemb   = (const float*)d_in[3];
    const float* eWih   = (const float*)d_in[4];
    const float* eWhh   = (const float*)d_in[5];
    const float* ebih   = (const float*)d_in[6];
    const float* ebhh   = (const float*)d_in[7];
    const float* dWih   = (const float*)d_in[8];
    const float* dWhh   = (const float*)d_in[9];
    const float* dbih   = (const float*)d_in[10];
    const float* dbhh   = (const float*)d_in[11];
    const float* dstart = (const float*)d_in[12];
    const float* eos    = (const float*)d_in[13];

    char* ws = (char*)d_ws;
    size_t off = 0;
    auto alloc = [&](size_t bytes) -> void* {
        void* p = ws + off;
        off = (off + bytes + 255) & ~(size_t)255;
        return p;
    };
    bf16_t* Wemb_b = (bf16_t*)alloc((size_t)Ee * Nn * 2);
    bf16_t* eWih_b = (bf16_t*)alloc((size_t)G4 * Ee * 2);
    bf16_t* eWhh_b = (bf16_t*)alloc((size_t)G4 * Hh * 2);
    bf16_t* dWih_b = (bf16_t*)alloc((size_t)G4 * Ee * 2);
    bf16_t* dWhh_b = (bf16_t*)alloc((size_t)G4 * Hh * 2);
    bf16_t* emb_b  = (bf16_t*)alloc((size_t)Bm * Nn * Ee * 2);
    float*  ext    = (float*) alloc((size_t)Bm * NP1 * Hh * 4);
    float*  hf     = (float*) alloc((size_t)Bm * Hh * 4);
    bf16_t* hb     = (bf16_t*)alloc((size_t)Bm * Hh * 2);
    float*  cf     = (float*) alloc((size_t)Bm * Hh * 4);
    float*  zf     = (float*) alloc((size_t)Bm * G4 * 4);
    bf16_t* dec_in = (bf16_t*)alloc((size_t)Bm * Ee * 2);
    float*  logits = (float*) alloc((size_t)Bm * NP1 * 4);
    unsigned char* mask = (unsigned char*)alloc((size_t)Bm * NP1);
    float*  loss   = (float*) alloc(256);

    // one-time prep: bf16 weight copies + state init
    f32_to_bf16<<<(Ee * Nn + 255) / 256, 256, 0, stream>>>(Wemb, Wemb_b, Ee * Nn);
    f32_to_bf16<<<(G4 * Ee + 255) / 256, 256, 0, stream>>>(eWih, eWih_b, G4 * Ee);
    f32_to_bf16<<<(G4 * Hh + 255) / 256, 256, 0, stream>>>(eWhh, eWhh_b, G4 * Hh);
    f32_to_bf16<<<(G4 * Ee + 255) / 256, 256, 0, stream>>>(dWih, dWih_b, G4 * Ee);
    f32_to_bf16<<<(G4 * Hh + 255) / 256, 256, 0, stream>>>(dWhh, dWhh_b, G4 * Hh);
    init_state<<<(Bm * Hh + 255) / 256, 256, 0, stream>>>(hf, hb, cf, mask, loss);

    // node embeddings: (B*N/16)*(E/16) = 32768 wave-tiles, 8 waves/block
    embed_gemm<<<(Bm * Nn / 16) * (Ee / 16) / 8, 256, 0, stream>>>(adj, Wemb_b, bemb, emb_b);
    init_eos<<<(Bm * Hh + 255) / 256, 256, 0, stream>>>(eos, ext);

    const int gemm_blocks = (Bm / 16) * (G4 / 16) / 8;   // 2048 tiles -> 256 blocks
    const int gate_blocks = (Bm * Hh + 255) / 256;

    // ---- encoder: 128 sequential steps ----
    for (int t = 0; t < Nn; ++t) {
        lstm_gemm_step<<<gemm_blocks, 256, 0, stream>>>(
            emb_b + (size_t)t * Ee, Nn * Ee, hb, eWih_b, eWhh_b, ebih, ebhh, zf);
        lstm_gate<<<gate_blocks, 256, 0, stream>>>(zf, cf, hf, hb, ext, t);
    }

    // ---- decoder: 129 sequential steps ----
    init_dec0<<<(Bm * Ee + 255) / 256, 256, 0, stream>>>(dstart, dec_in);
    for (int t = 0; t < Tt; ++t) {
        if (t > 0)
            gather_dec<<<(Bm * Ee + 255) / 256, 256, 0, stream>>>(emb_b, tgt, t - 1, dec_in);
        lstm_gemm_step<<<gemm_blocks, 256, 0, stream>>>(
            dec_in, Ee, hb, dWih_b, dWhh_b, dbih, dbhh, zf);
        lstm_gate<<<gate_blocks, 256, 0, stream>>>(zf, cf, hf, hb, nullptr, 0);
        attn_logits<<<(Bm * NP1) / 8, 256, 0, stream>>>(ext, hf, logits);
        ce_loss_step<<<Bm, 32, 0, stream>>>(logits, mask, tgt, t, loss);
    }

    finalize<<<1, 32, 0, stream>>>(loss, (float*)d_out);
}
